// MultiHeadedAttention_28295244546119
// MI455X (gfx1250) — matrix operand order repaired and checked by reading, not compiled, not run on hardware
//
#include <hip/hip_runtime.h>
#include <hip/hip_bf16.h>

typedef _Float16 half_t;
typedef __attribute__((ext_vector_type(16))) _Float16 v16h;
typedef __attribute__((ext_vector_type(8)))  float    v8f;
typedef __attribute__((ext_vector_type(4)))  unsigned u32x4;
typedef __attribute__((ext_vector_type(8)))  unsigned u32x8;

#define NB   2
#define SEQ  2048
#define EMB  1024
#define NH   16
#define HD   64
#define MTOT (NB*SEQ)   // 4096 tokens

// TDM via inline asm (portable across toolchains; bypasses the clang builtin
// whose arity differs between ROCm 7.2 and the therock lane).
#define USE_TDM 1

__device__ __forceinline__ void wait_tensor0() {
#if defined(__has_builtin) && __has_builtin(__builtin_amdgcn_s_wait_tensorcnt)
  __builtin_amdgcn_s_wait_tensorcnt(0);
#else
  asm volatile("s_wait_tensorcnt 0x0" ::: "memory");
#endif
}

#if USE_TDM
// DMA a 2D f16 tile (rows x 64 elems) from a row-major tensor (row stride
// `stride_elems`) into LDS at byte offset `lds_addr`, padding each 128B row
// by 16B so the LDS image has a 72-half row stride (matches [rows][72]).
// D# packing per CDNA5 ISA 8.3/8.4 (group0: count/lds/global/type, group1:
// data_size/pad/dims/strides).  2D tensor -> 2 descriptor groups (VADDR2/3
// NULL).  Issue from ONE wave; EXEC is ignored; tracked by TENSORcnt.
__device__ __forceinline__ void tdm_load_2d(unsigned lds_addr, const half_t* gptr,
                                            unsigned rows, unsigned stride_elems) {
  unsigned long long ga = (unsigned long long)(size_t)gptr;
  u32x4 g0;
  g0[0] = 1u;                                             // count=1 (valid)
  g0[1] = lds_addr;                                       // LDS byte address
  g0[2] = (unsigned)(ga & 0xffffffffull);                 // global_addr[31:0]
  g0[3] = (unsigned)((ga >> 32) & 0x1ffffffull) | (2u << 30); // addr[56:32]|type=2
  const unsigned td0 = 64, td1 = rows, t0 = 64, t1 = rows;
  u32x8 g1;
  // data_size=1 (2B) | pad_enable | pad_interval=4 (32 DWORDs) | pad_amount=3 (4 DWORDs)
  g1[0] = (1u << 16) | (1u << 20) | (4u << 22) | (3u << 25);
  g1[1] = (td0 & 0xffffu) << 16;                          // tensor_dim0[15:0]
  g1[2] = ((td0 >> 16) & 0xffffu) | ((td1 & 0xffffu) << 16);
  g1[3] = ((td1 >> 16) & 0xffffu) | ((t0 & 0xffffu) << 16);   // tile_dim0
  g1[4] = (t1 & 0xffffu);                                 // tile_dim1 (tile_dim2=0)
  g1[5] = stride_elems;                                   // tensor_dim0_stride[31:0]
  g1[6] = 0u;
  g1[7] = 0u;
  asm volatile("tensor_load_to_lds %0, %1" :: "s"(g0), "s"(g1) : "memory");
}
#endif

__device__ __forceinline__ v8f wmma16(v16h a, v16h b, v8f c) {
  return __builtin_amdgcn_wmma_f32_16x16x32_f16(false, a, false, b, (short)0, c,
                                                false, false);
}

// A fragment (16 rows x 32 K, f16) from row-major LDS tile; ld = row stride (halves).
__device__ __forceinline__ v16h load_frag_a(const half_t* p, int ld) {
  int lane = threadIdx.x & 31;
  int grp  = lane >> 4;
  const half_t* row = p + (lane & 15) * ld;
  v16h a;
#pragma unroll
  for (int v = 0; v < 8; ++v) {
    int kb = (v < 4) ? (8*grp + 2*v) : (8 + 8*grp + 2*v);
    a[2*v]   = row[kb];
    a[2*v+1] = row[kb + 1];
  }
  return a;
}

// B fragment (32 K x 16 N) from LDS tile stored as B^T row-major (16 N rows x 32 K).
__device__ __forceinline__ v16h load_frag_b(const half_t* p, int ld) {
  int lane = threadIdx.x & 31;
  int grp  = lane >> 4;
  const half_t* row = p + (lane & 15) * ld;
  v16h b;
#pragma unroll
  for (int v = 0; v < 8; ++v) {
    int kb = (v < 4) ? (8*grp + 2*v) : (8 + 8*grp + 2*v);
    b[2*v]   = row[kb];
    b[2*v+1] = row[kb + 1];
  }
  return b;
}

// ---------------------------------------------------------------------------
// Kernel 1: Q/K/V projection.  Y = X*W^T + b, f16 outputs.
//   z=0 -> Qh (N,H,S,D);  z=1 -> Kh (N,H,S,D);  z=2 -> Vt (N,H,D,S) TRANSPOSED
// Block tile 64x128xK32, 8 waves in 2x4 grid, 32x32 per wave (4 WMMAs/step).
// ---------------------------------------------------------------------------
__global__ __launch_bounds__(256)
void proj_heads_kernel(const float* __restrict__ Xq, const float* __restrict__ Xk,
                       const float* __restrict__ Xv,
                       const float* __restrict__ Wq, const float* __restrict__ Wk,
                       const float* __restrict__ Wv,
                       const float* __restrict__ bq, const float* __restrict__ bk,
                       const float* __restrict__ bv,
                       half_t* __restrict__ Qh, half_t* __restrict__ Kh,
                       half_t* __restrict__ Vt) {
  const float* X; const float* W; const float* bias; half_t* Out;
  if (blockIdx.z == 0)      { X = Xq; W = Wq; bias = bq; Out = Qh; }
  else if (blockIdx.z == 1) { X = Xk; W = Wk; bias = bk; Out = Kh; }
  else                      { X = Xv; W = Wv; bias = bv; Out = Vt; }
  const bool transpose_out = (blockIdx.z == 2);

  __shared__ half_t As[64][40];    // 64 tokens x 32 K
  __shared__ half_t Bs[128][40];   // 128 out-features x 32 K (B^T layout)

  int tid  = threadIdx.x;
  int wid  = tid >> 5, lane = tid & 31;
  int wr   = wid >> 2;             // 0..1 : 32-row block
  int wc   = wid & 3;              // 0..3 : 32-col block
  int nlo  = lane & 15, mhi = (lane >> 4) * 8;
  int m0   = blockIdx.x * 64;
  int n0   = blockIdx.y * 128;

  v8f c00 = {}, c01 = {}, c10 = {}, c11 = {};
  for (int k0 = 0; k0 < EMB; k0 += 32) {
#pragma unroll
    for (int j = 0; j < 8; ++j) {
      int linear = j*256 + tid;
      int r = linear >> 5, c = linear & 31;
      As[r][c] = (half_t)X[(size_t)(m0 + r)*EMB + k0 + c];
    }
#pragma unroll
    for (int j = 0; j < 16; ++j) {
      int linear = j*256 + tid;
      int r = linear >> 5, c = linear & 31;
      Bs[r][c] = (half_t)W[(size_t)(n0 + r)*EMB + k0 + c];
    }
    __syncthreads();
    v16h a0 = load_frag_a(&As[wr*32][0], 40);
    v16h a1 = load_frag_a(&As[wr*32 + 16][0], 40);
    v16h b0 = load_frag_b(&Bs[wc*32][0], 40);
    v16h b1 = load_frag_b(&Bs[wc*32 + 16][0], 40);
    c00 = wmma16(a0, b0, c00);
    c01 = wmma16(a0, b1, c01);
    c10 = wmma16(a1, b0, c10);
    c11 = wmma16(a1, b1, c11);
    __syncthreads();
  }

#pragma unroll
  for (int half = 0; half < 2; ++half) {
    v8f* cr0 = half ? &c10 : &c00;
    v8f* cr1 = half ? &c11 : &c01;
#pragma unroll
    for (int r = 0; r < 8; ++r) {
      int m = m0 + wr*32 + half*16 + r + mhi;   // global token
      int b = m >> 11;                          // / SEQ
      int s = m & (SEQ - 1);
      int col0 = n0 + wc*32 + nlo;
      int col1 = col0 + 16;
      float v0 = (*cr0)[r] + bias[col0];
      float v1 = (*cr1)[r] + bias[col1];
      if (!transpose_out) {
        Out[(((size_t)b*NH + (col0 >> 6))*SEQ + s)*HD + (col0 & 63)] = (half_t)v0;
        Out[(((size_t)b*NH + (col1 >> 6))*SEQ + s)*HD + (col1 & 63)] = (half_t)v1;
      } else {
        Out[(((size_t)b*NH + (col0 >> 6))*HD + (col0 & 63))*SEQ + s] = (half_t)v0;
        Out[(((size_t)b*NH + (col1 >> 6))*HD + (col1 & 63))*SEQ + s] = (half_t)v1;
      }
    }
  }
}

// ---------------------------------------------------------------------------
// Kernel 2: causal flash attention per (n,h).  grid = (SEQ/64, NB*NH), 256 thr.
// Q resident; K/V double-buffered and streamed via TDM (DMA overlapped with
// compute; TENSORcnt waited only at iteration end).  P tile aliases the dead
// current K buffer to keep static LDS under 64KB.
// ---------------------------------------------------------------------------
__global__ __launch_bounds__(256)
void attn_kernel(const half_t* __restrict__ Qh, const half_t* __restrict__ Kh,
                 const half_t* __restrict__ Vt, half_t* __restrict__ Ctx) {
  __shared__ float  Ss [64][68];       // scores f32
  __shared__ float  red[4][64];
  __shared__ float  row_m[64], row_l[64], row_alpha[64];
  __shared__ half_t Qs  [64][72];      // [s][d]
  __shared__ half_t Kbuf[2][64][72];   // [t][d]  (B^T for Q*K^T); also hosts P
  __shared__ half_t Vbuf[2][64][72];   // [d][t]  (B^T for P*V)

  int tid  = threadIdx.x;
  int wid  = tid >> 5, lane = tid & 31;
  int wr   = wid >> 1, wc = wid & 1;         // 4x2 wave grid, 16x32 per wave
  int nlo  = lane & 15, mhi = (lane >> 4) * 8;
  int qb   = blockIdx.x;
  size_t base = (size_t)blockIdx.y * SEQ * HD;   // same flat size for K and Vt
  int bn = blockIdx.y / NH, hh = blockIdx.y % NH;
  const float scale = 0.125f;                 // 1/sqrt(64)

  // ---- preamble: stage Q and tile 0 of K/V ----
#if USE_TDM
  if (wid == 0) {
    tdm_load_2d((unsigned)(size_t)&Qs[0][0],      Qh + base + (size_t)(qb*64)*HD, 64, HD);
    tdm_load_2d((unsigned)(size_t)&Kbuf[0][0][0], Kh + base,                      64, HD);
    tdm_load_2d((unsigned)(size_t)&Vbuf[0][0][0], Vt + base,                      64, SEQ);
    wait_tensor0();
  }
#else
#pragma unroll
  for (int j = 0; j < 16; ++j) {
    int linear = j*256 + tid;
    int r = linear >> 6, c = linear & 63;
    Qs[r][c]      = Qh[base + (size_t)(qb*64 + r)*HD + c];
    Kbuf[0][r][c] = Kh[base + (size_t)r*HD + c];
    Vbuf[0][r][c] = Vt[base + (size_t)r*SEQ + c];
  }
#endif
  if (tid < 64) { row_m[tid] = -1e30f; row_l[tid] = 0.f; }
  v8f o0 = {}, o1 = {};
  __syncthreads();

  int row = tid & 63, part = tid >> 6;
  int cur = 0;

  for (int kb = 0; kb <= qb; ++kb) {
    int nxt = cur ^ 1;
#if USE_TDM
    // Prefetch next K/V tiles into the alternate buffers (overlaps with compute).
    if (kb < qb && wid == 0) {
      tdm_load_2d((unsigned)(size_t)&Kbuf[nxt][0][0],
                  Kh + base + (size_t)((kb + 1)*64)*HD, 64, HD);
      tdm_load_2d((unsigned)(size_t)&Vbuf[nxt][0][0],
                  Vt + base + (size_t)((kb + 1)*64),    64, SEQ);
    }
#endif

    // ---- S = Q * K^T ----
    v8f s0 = {}, s1 = {};
#pragma unroll
    for (int kk = 0; kk < 64; kk += 32) {
      v16h a  = load_frag_a(&Qs[wr*16][kk], 72);
      v16h b0 = load_frag_b(&Kbuf[cur][wc*32][kk], 72);
      v16h b1 = load_frag_b(&Kbuf[cur][wc*32 + 16][kk], 72);
      s0 = wmma16(a, b0, s0);
      s1 = wmma16(a, b1, s1);
    }
#pragma unroll
    for (int r = 0; r < 8; ++r) {
      int m  = wr*16 + r + mhi;
      int sg = qb*64 + m;
      int t0 = kb*64 + wc*32 + nlo;
      int t1 = t0 + 16;
      Ss[m][wc*32 + nlo]      = (t0 <= sg) ? s0[r]*scale : -1e12f;
      Ss[m][wc*32 + nlo + 16] = (t1 <= sg) ? s1[r]*scale : -1e12f;
    }
    __syncthreads();   // scores visible; K[cur] fully consumed -> reusable for P

    // ---- online softmax (4 threads per row, 16 cols each) ----
    half_t (*Ps)[72] = Kbuf[cur];     // P tile aliases the dead K buffer
    float pm = -1e30f;
#pragma unroll
    for (int j = 0; j < 16; ++j) pm = fmaxf(pm, Ss[row][part*16 + j]);
    red[part][row] = pm;
    __syncthreads();
    if (tid < 64) {
      float bm  = fmaxf(fmaxf(red[0][tid], red[1][tid]),
                        fmaxf(red[2][tid], red[3][tid]));
      float old = row_m[tid];
      float nm  = fmaxf(old, bm);
      row_alpha[tid] = __expf(old - nm);
      row_m[tid]     = nm;
    }
    __syncthreads();
    float nm = row_m[row];
    float psum = 0.f;
#pragma unroll
    for (int j = 0; j < 16; ++j) {
      int c = part*16 + j;
      float p = __expf(Ss[row][c] - nm);
      Ps[row][c] = (half_t)p;
      psum += p;
    }
    red[part][row] = psum;
    __syncthreads();
    if (tid < 64)
      row_l[tid] = row_l[tid]*row_alpha[tid]
                 + red[0][tid] + red[1][tid] + red[2][tid] + red[3][tid];
    __syncthreads();

    // ---- O = alpha*O + P*V ----
#pragma unroll
    for (int r = 0; r < 8; ++r) {
      float al = row_alpha[wr*16 + r + mhi];
      o0[r] *= al; o1[r] *= al;
    }
#pragma unroll
    for (int kk = 0; kk < 64; kk += 32) {
      v16h a  = load_frag_a(&Ps[wr*16][kk], 72);
      v16h b0 = load_frag_b(&Vbuf[cur][wc*32][kk], 72);
      v16h b1 = load_frag_b(&Vbuf[cur][wc*32 + 16][kk], 72);
      o0 = wmma16(a, b0, o0);
      o1 = wmma16(a, b1, o1);
    }

#if USE_TDM
    if (wid == 0) wait_tensor0();     // next K/V tiles landed
    __syncthreads();
#else
    __syncthreads();                  // all reads of cur buffers done
    if (kb < qb) {
#pragma unroll
      for (int j = 0; j < 16; ++j) {
        int linear = j*256 + tid;
        int r = linear >> 6, c = linear & 63;
        Kbuf[nxt][r][c] = Kh[base + (size_t)((kb + 1)*64 + r)*HD + c];
        Vbuf[nxt][r][c] = Vt[base + (size_t)r*SEQ + (kb + 1)*64 + c];
      }
      __syncthreads();
    }
#endif
    cur = nxt;
  }

  // ---- normalize and write ctx as (N,S,E) f16 ----
#pragma unroll
  for (int r = 0; r < 8; ++r) {
    int m = wr*16 + r + mhi;
    int s = qb*64 + m;
    float linv = 1.f / row_l[m];
    int d0 = wc*32 + nlo, d1 = d0 + 16;
    Ctx[((size_t)bn*SEQ + s)*EMB + hh*HD + d0] = (half_t)(o0[r]*linv);
    Ctx[((size_t)bn*SEQ + s)*EMB + hh*HD + d1] = (half_t)(o1[r]*linv);
  }
}

// ---------------------------------------------------------------------------
// Kernel 3: output projection.  out(f32) = ctx(f16) * Wp^T + bp
// Block tile 64x128xK32, 32x32 per wave.  grid = (MTOT/64, EMB/128), 256 thr.
// ---------------------------------------------------------------------------
__global__ __launch_bounds__(256)
void out_proj_kernel(const half_t* __restrict__ Ctx, const float* __restrict__ Wp,
                     const float* __restrict__ bp, float* __restrict__ Out) {
  __shared__ half_t As[64][40];
  __shared__ half_t Bs[128][40];
  int tid = threadIdx.x;
  int wid = tid >> 5, lane = tid & 31;
  int wr  = wid >> 2, wc = wid & 3;
  int nlo = lane & 15, mhi = (lane >> 4) * 8;
  int m0  = blockIdx.x * 64, n0 = blockIdx.y * 128;

  v8f c00 = {}, c01 = {}, c10 = {}, c11 = {};
  for (int k0 = 0; k0 < EMB; k0 += 32) {
#pragma unroll
    for (int j = 0; j < 8; ++j) {
      int linear = j*256 + tid;
      int r = linear >> 5, c = linear & 31;
      As[r][c] = Ctx[(size_t)(m0 + r)*EMB + k0 + c];
    }
#pragma unroll
    for (int j = 0; j < 16; ++j) {
      int linear = j*256 + tid;
      int r = linear >> 5, c = linear & 31;
      Bs[r][c] = (half_t)Wp[(size_t)(n0 + r)*EMB + k0 + c];
    }
    __syncthreads();
    v16h a0 = load_frag_a(&As[wr*32][0], 40);
    v16h a1 = load_frag_a(&As[wr*32 + 16][0], 40);
    v16h b0 = load_frag_b(&Bs[wc*32][0], 40);
    v16h b1 = load_frag_b(&Bs[wc*32 + 16][0], 40);
    c00 = wmma16(a0, b0, c00);
    c01 = wmma16(a0, b1, c01);
    c10 = wmma16(a1, b0, c10);
    c11 = wmma16(a1, b1, c11);
    __syncthreads();
  }
#pragma unroll
  for (int half = 0; half < 2; ++half) {
    v8f* cr0 = half ? &c10 : &c00;
    v8f* cr1 = half ? &c11 : &c01;
#pragma unroll
    for (int r = 0; r < 8; ++r) {
      int m = m0 + wr*32 + half*16 + r + mhi;
      int col0 = n0 + wc*32 + nlo, col1 = col0 + 16;
      Out[(size_t)m*EMB + col0] = (*cr0)[r] + bp[col0];
      Out[(size_t)m*EMB + col1] = (*cr1)[r] + bp[col1];
    }
  }
}

extern "C" void kernel_launch(void* const* d_in, const int* in_sizes, int n_in,
                              void* d_out, int out_size, void* d_ws, size_t ws_size,
                              hipStream_t stream) {
  (void)in_sizes; (void)n_in; (void)out_size; (void)ws_size;
  const float* key   = (const float*)d_in[0];
  const float* value = (const float*)d_in[1];
  const float* query = (const float*)d_in[2];
  // d_in[3] is the causal tril mask; applied analytically in attn_kernel.
  const float* Wk = (const float*)d_in[4];
  const float* bk = (const float*)d_in[5];
  const float* Wq = (const float*)d_in[6];
  const float* bq = (const float*)d_in[7];
  const float* Wv = (const float*)d_in[8];
  const float* bv = (const float*)d_in[9];
  const float* Wp = (const float*)d_in[10];
  const float* bp = (const float*)d_in[11];
  float* out = (float*)d_out;

  size_t per = (size_t)NB * NH * SEQ * HD;   // 4M f16 elements each
  half_t* Qh  = (half_t*)d_ws;
  half_t* Kh  = Qh + per;
  half_t* Vt  = Kh + per;                    // (N,H,D,S) transposed
  half_t* Ctx = Vt + per;                    // (N,S,E) f16

  proj_heads_kernel<<<dim3(MTOT/64, EMB/128, 3), 256, 0, stream>>>(
      query, key, value, Wq, Wk, Wv, bq, bk, bv, Qh, Kh, Vt);
  attn_kernel<<<dim3(SEQ/64, NB*NH), 256, 0, stream>>>(Qh, Kh, Vt, Ctx);
  out_proj_kernel<<<dim3(MTOT/64, EMB/128), 256, 0, stream>>>(Ctx, Wp, bp, out);
}
